// Linear_Int4_73821897883660
// MI455X (gfx1250) — matrix operand order repaired and checked by reading, not compiled, not run on hardware
//
#include <hip/hip_runtime.h>
#include <hip/hip_bf16.h>

typedef __attribute__((ext_vector_type(16))) __bf16 v16bf;
typedef __attribute__((ext_vector_type(2)))  __bf16 v2bf;
typedef __attribute__((ext_vector_type(8)))  float  v8f;

#define BM 128
#define BN 128
#define BK 64
#define LDAP 36   // padded row stride (u32 bf16-pairs) for A tile (BK/2=32 -> 36; 16B aligned, bank-spread)
#define LDBP 36   // padded row stride (u32 bf16-pairs) for B^T tile

__device__ __forceinline__ unsigned permb32(unsigned s0, unsigned s1, unsigned sel) {
#if __has_builtin(__builtin_amdgcn_perm)
  return __builtin_amdgcn_perm(s0, s1, sel);
#else
  unsigned r = 0;
  for (int i = 0; i < 4; ++i) {
    unsigned s = (sel >> (8 * i)) & 0xFFu;
    unsigned byte = (s < 4u) ? ((s1 >> (8 * s)) & 0xFFu)
                   : (s < 8u) ? ((s0 >> (8 * (s - 4u))) & 0xFFu) : 0u;
    r |= byte << (8 * i);
  }
  return r;
#endif
}

// fp32 pair -> packed bf16 pair (round-half-up): 2 adds + 1 perm.
__device__ __forceinline__ unsigned packbf2(float a, float b) {
  unsigned ua = __builtin_bit_cast(unsigned, a) + 0x8000u;
  unsigned ub = __builtin_bit_cast(unsigned, b) + 0x8000u;
  return permb32(ub, ua, 0x07060302u);  // {ub.hi16, ua.hi16}
}

// Nibble pair p of word -> packed bf16 pair holding exact (q_even, q_odd).
// Magic: bf16(128+n) == 0x4300|n (exact for n in 0..15); then v_pk_add_bf16(-128).
__device__ __forceinline__ unsigned nib2bf2(unsigned o, unsigned e, unsigned p) {
  unsigned sel = 0x0C000C00u | p | ((4u + p) << 16);          // byte0=e[p], byte2=o[p], others 0
  unsigned t = permb32(o, e, sel) | 0x43004300u;               // {bf16(128+odd), bf16(128+even)}
  const v2bf m128 = {(__bf16)-128.0f, (__bf16)-128.0f};
  v2bf r = __builtin_bit_cast(v2bf, t) + m128;                 // v_pk_add_bf16 -> exact {q_odd, q_even}
  return __builtin_bit_cast(unsigned, r);
}

union FragAB { v16bf v; uint4 q[2]; };

__global__ __launch_bounds__(256)
void w4a16_wmma_kernel(const float* __restrict__ x,
                       const int*   __restrict__ qw,
                       const float* __restrict__ scales,
                       const float* __restrict__ zeros,
                       const float* __restrict__ bias,
                       float*       __restrict__ out,
                       int M, int N, int K)
{
  __shared__ __align__(16) unsigned ldsA[2 * BM * LDAP];   // double-buffered x tile   [buf][m][k2]
  __shared__ __align__(16) unsigned ldsB[2 * BN * LDBP];   // double-buffered q tile^T [buf][n][k2]
  __shared__ float ldsRS[BM * 16];                         // rowsum partials
  __shared__ float ldsRSf[BM];                             // final rowsums

  const int tid  = threadIdx.x;
  const int lane = tid & 31;
  const int wave = tid >> 5;
  const int hi   = lane >> 4;     // lane half (WMMA layout selector)
  const int lo16 = lane & 15;
  const int wm   = wave >> 1;     // 0..3 -> 32-row slab
  const int wn   = wave & 1;      // 0..1 -> 64-col slab

  const int mTile = blockIdx.y * BM;
  const int nTile = blockIdx.x * BN;

  // A-loader: 256 threads cover 128 rows x 16 float4 (K-tile = 64 floats)
  const int aCol4 = tid & 15;
  const int aRow0 = tid >> 4;
  // B-loader: 256 threads cover 8 q-rows x 128 cols of int32 words
  const int bN  = tid & 127;
  const int bQ0 = (tid >> 7) * 4;

  const int    KT       = K / BK;
  const size_t strideX4 = (size_t)(K >> 2);
  const float4* xv = (const float4*)x;

  v8f    acc[2][4] = {};
  float2 rs2[8];
  #pragma unroll
  for (int i = 0; i < 8; ++i) { rs2[i].x = 0.f; rs2[i].y = 0.f; }
  float4 ga[8];
  unsigned gb[4];

  // ---- prologue: stage tile kt = 0 into registers ----
  #pragma unroll
  for (int i = 0; i < 8; ++i)
    ga[i] = xv[(size_t)(mTile + aRow0 + i * 16) * strideX4 + aCol4];
  #pragma unroll
  for (int w = 0; w < 4; ++w)
    gb[w] = (unsigned)qw[(size_t)(bQ0 + w) * N + (nTile + bN)];

  #pragma unroll 2   // match ping-pong period: kills register-rotation movs
  for (int kt = 0; kt < KT; ++kt) {
    unsigned* lA = &ldsA[(kt & 1) * (BM * LDAP)];
    unsigned* lB = &ldsB[(kt & 1) * (BN * LDBP)];

    // ---- store staged tile (convert fp32->bf16, int4->bf16) ----
    #pragma unroll
    for (int i = 0; i < 8; ++i) {
      float4 f = ga[i];
      // two v_pk_add_f32 on naturally adjacent pairs (f.xy + f.zw), no shuffles
      rs2[i].x += (f.x + f.z);
      rs2[i].y += (f.y + f.w);
      uint2 p;
      p.x = packbf2(f.x, f.y);
      p.y = packbf2(f.z, f.w);
      *(uint2*)&lA[(aRow0 + i * 16) * LDAP + aCol4 * 2] = p;
    }
    #pragma unroll
    for (int w = 0; w < 4; ++w) {
      unsigned word = gb[w];
      unsigned e = word & 0x0F0F0F0Fu;
      unsigned o = (word >> 4) & 0x0F0F0F0Fu;
      uint4 st;
      st.x = nib2bf2(o, e, 0u);
      st.y = nib2bf2(o, e, 1u);
      st.z = nib2bf2(o, e, 2u);
      st.w = nib2bf2(o, e, 3u);
      *(uint4*)&lB[bN * LDBP + (bQ0 + w) * 4] = st;
    }

    // ---- issue next tile's global loads (latency overlaps compute below) ----
    if (kt + 1 < KT) {
      #pragma unroll
      for (int i = 0; i < 8; ++i)
        ga[i] = xv[(size_t)(mTile + aRow0 + i * 16) * strideX4 + (size_t)(kt + 1) * 16 + aCol4];
      #pragma unroll
      for (int w = 0; w < 4; ++w)
        gb[w] = (unsigned)qw[(size_t)((kt + 1) * 8 + bQ0 + w) * N + (nTile + bN)];
    }

    __syncthreads();   // single barrier per K-tile (ping-pong buffers)

    // ---- compute: two 16x16x32 bf16 k-steps ----
    #pragma unroll
    for (int s = 0; s < 2; ++s) {
      FragAB a[2], b[4];
      #pragma unroll
      for (int tm = 0; tm < 2; ++tm) {
        // ISA 16-bit A 16x32 layout: VGPR j holds K-pair k2 = (j&4)*2 + hi*4 + (j&3)
        int mRow = wm * 32 + tm * 16 + lo16;
        const unsigned* base = &lA[mRow * LDAP + s * 16 + hi * 4];
        a[tm].q[0] = *(const uint4*)(base);
        a[tm].q[1] = *(const uint4*)(base + 8);
      }
      #pragma unroll
      for (int tn = 0; tn < 4; ++tn) {
        // ISA 16-bit B 32x16 layout: VGPR j holds K-pair k2 = (j&4)*2 + hi*4 + (j&3) at column lane&15
        int nCol = wn * 64 + tn * 16 + lo16;
        const unsigned* base = &lB[nCol * LDBP + s * 16 + hi * 4];
        b[tn].q[0] = *(const uint4*)(base);
        b[tn].q[1] = *(const uint4*)(base + 8);
      }
      #pragma unroll
      for (int tm = 0; tm < 2; ++tm)
        #pragma unroll
        for (int tn = 0; tn < 4; ++tn)
          acc[tm][tn] = __builtin_amdgcn_wmma_f32_16x16x32_bf16(
              false, a[tm].v, false, b[tn].v, (short)0, acc[tm][tn], false, false);
    }
  }

  // ---- reduce rowsums (zero-point correction term) ----
  #pragma unroll
  for (int i = 0; i < 8; ++i)
    ldsRS[(aRow0 + i * 16) * 16 + aCol4] = rs2[i].x + rs2[i].y;
  __syncthreads();
  if (tid < BM) {
    float s = 0.f;
    #pragma unroll
    for (int c = 0; c < 16; ++c) s += ldsRS[tid * 16 + c];
    ldsRSf[tid] = s;
  }
  __syncthreads();

  // ---- epilogue: out = scale*acc - scale*zero*rowsum + bias (fp32) ----
  #pragma unroll
  for (int tn = 0; tn < 4; ++tn) {
    int   ng  = nTile + wn * 64 + tn * 16 + lo16;
    float sc  = scales[ng];
    float zp  = zeros[ng];
    float bz  = bias[ng];
    float szp = sc * zp;
    #pragma unroll
    for (int tm = 0; tm < 2; ++tm) {
      #pragma unroll
      for (int v = 0; v < 8; ++v) {
        // ISA 32-bit C/D 16x16 layout: VGPR v holds M = v + 8*hi, N = lane&15
        int   mLoc = wm * 32 + tm * 16 + v + 8 * hi;
        float o = sc * acc[tm][tn][v] - szp * ldsRSf[mLoc] + bz;
        out[(size_t)(mTile + mLoc) * N + ng] = o;
      }
    }
  }
}

extern "C" void kernel_launch(void* const* d_in, const int* in_sizes, int n_in,
                              void* d_out, int out_size, void* d_ws, size_t ws_size,
                              hipStream_t stream) {
  const float* x      = (const float*)d_in[0];
  const int*   qw     = (const int*)d_in[1];
  const float* scales = (const float*)d_in[2];
  const float* zeros  = (const float*)d_in[3];
  const float* bias   = (const float*)d_in[4];
  float*       out    = (float*)d_out;

  const int N = in_sizes[2];
  const int K = (int)(((long long)in_sizes[1] / N) * 8);
  const int M = (int)((long long)in_sizes[0] / K);

  dim3 grid((unsigned)(N / BN), (unsigned)(M / BM));
  dim3 block(256);
  w4a16_wmma_kernel<<<grid, block, 0, stream>>>(x, qw, scales, zeros, bias, out, M, N, K);
}
